// Roenet_32830730010847
// MI455X (gfx1250) — matrix operand order repaired and checked by reading, not compiled, not run on hardware
//
#include <hip/hip_runtime.h>
#include <hip/hip_bf16.h>

typedef __attribute__((ext_vector_type(16))) _Float16 v16h;
typedef __attribute__((ext_vector_type(8)))  _Float16 h8;
typedef __attribute__((ext_vector_type(8)))  float    v8f;

#define NZ     8212      // spatial length of z
#define NP     8213      // extended positions (e[x] = (u[x-1],u[x]))
#define NPAD   8224      // row stride for lam/L arrays
#define WBUF   128       // tile width (valid cols)
#define NCOL   130       // WBUF + 2 guard columns
#define CSTR   64        // channel stride (halves) in x-major LDS layout
#define TOUT   118       // valid outputs per tile (WBUF - 2*5 halo)
#define NT     8         // WBUF/16 N-tiles
#define NWAVES 8
#define NTILES 70        // ceil(8213/118)
#define NBATCH 16

struct StackArgs {
  const _Float16* w1[6];    // k=3 blocks: [dx][Cout][Kd32(Cin)] ; blk5: [Cout][Kd32]
  const _Float16* w2[6];    // [Cout][Kd32(Cout)]
  const _Float16* wsc[3];   // shortcut 1x1 convs for blocks 0,2,3: [Cout][Kd32(Cin)]
  const _Float16* fw;
  const float* b1[6];
  const float* b2[6];
  const float* bsc[3];
  const float* fb;
};

__device__ static inline int clampi(int v, int lo, int hi) {
  return v < lo ? lo : (v > hi ? hi : v);
}

// One GEMM accumulation, x-major LDS activations, compile-time K depth.
// A (16x32 f16): lane<16 -> K {k0+0..7, k0+16..23}; lane>=16 -> {k0+8..15, k0+24..31}; M = lane%16.
// B (32x16 f16): same K order as A: lane<16 -> K {k0+0..7, k0+16..23}; lane>=16 -> {k0+8..15, k0+24..31}; N = lane%16.
// S element [K][col] lives at S[col*CSTR + K] -> two contiguous 16B LDS loads per chunk.
template<int KD32>
__device__ static inline void accum_gemm(v8f& acc, const _Float16* __restrict__ W,
                                         const _Float16* S, int colbase, int m, int lane) {
  const int row = lane & 15;
  const int hiv = (lane >> 4) & 1;
  const _Float16* wrow = W + (size_t)(m * 16 + row) * KD32 + hiv * 8;
  const _Float16* srow = S + colbase * CSTR + hiv * 8;
#pragma unroll
  for (int k0 = 0; k0 < KD32; k0 += 32) {
    h8 alo = *(const h8*)(wrow + k0);
    h8 ahi = *(const h8*)(wrow + k0 + 16);
    h8 blo = *(const h8*)(srow + k0);
    h8 bhi = *(const h8*)(srow + k0 + 16);
    v16h a, bv;
#pragma unroll
    for (int e = 0; e < 8; ++e) {
      a[e] = alo[e]; a[e + 8] = ahi[e];
      bv[e] = blo[e]; bv[e + 8] = bhi[e];
    }
    acc = __builtin_amdgcn_wmma_f32_16x16x32_f16(false, a, false, bv, (short)0, acc, false, false);
  }
}

// Store one 16x16 D-tile: ReLU (v_max) + pack, 8 consecutive channels -> one 16B LDS store.
__device__ static inline void store_tile(_Float16* D, const v8f& acc, int co0, int xc) {
  h8 r;
#pragma unroll
  for (int j = 0; j < 8; ++j) r[j] = (_Float16)fmaxf(acc[j], 0.f);
  *(h8*)(D + (size_t)(xc + 1) * CSTR + co0) = r;
}

// h = relu(conv3(S, W) + bias) -> D ; W layout [dx][COUT][KD32C]
template<int COUT, int KD32C>
__device__ static inline void conv3_pass(const _Float16* __restrict__ W, const float* __restrict__ bias,
                                         const _Float16* S, _Float16* D) {
  const int lane = threadIdx.x & 31, wave = threadIdx.x >> 5;
  const int hi8 = ((lane >> 4) & 1) * 8;
#pragma unroll
  for (int t = wave; t < (COUT / 16) * NT; t += NWAVES) {
    int m = t / NT, n = t % NT;
    int xb = n * 16 + (lane & 15);
    int co0 = m * 16 + hi8;
    v8f acc;
#pragma unroll
    for (int j = 0; j < 8; ++j) acc[j] = bias[co0 + j];
#pragma unroll
    for (int dx = 0; dx < 3; ++dx)   // tap dx reads column x+dx-1 -> buffer col xb+dx
      accum_gemm<KD32C>(acc, W + (size_t)dx * COUT * KD32C, S, xb + dx, m, lane);
    store_tile(D, acc, co0, xb);
  }
}

// h = relu(conv1(S, W) + bias) -> D
template<int COUT, int KD32>
__device__ static inline void conv1_pass(const _Float16* __restrict__ W, const float* __restrict__ bias,
                                         const _Float16* S, _Float16* D) {
  const int lane = threadIdx.x & 31, wave = threadIdx.x >> 5;
  const int hi8 = ((lane >> 4) & 1) * 8;
#pragma unroll
  for (int t = wave; t < (COUT / 16) * NT; t += NWAVES) {
    int m = t / NT, n = t % NT;
    int xb = n * 16 + (lane & 15);
    int co0 = m * 16 + hi8;
    v8f acc;
#pragma unroll
    for (int j = 0; j < 8; ++j) acc[j] = bias[co0 + j];
    accum_gemm<KD32>(acc, W, S, xb + 1, m, lane);
    store_tile(D, acc, co0, xb);
  }
}

// y = relu(conv1(H,W2)+b2 + shortcut) -> D. SCMODE: 1=identity(X), 2=conv1(X,Ws)+bs
template<int COUT, int KD32H, int SCMODE, int KD32X>
__device__ static inline void resout_pass(const _Float16* __restrict__ W2, const float* __restrict__ b2,
                                          const _Float16* H,
                                          const _Float16* __restrict__ Ws, const float* __restrict__ bs,
                                          const _Float16* X, _Float16* D) {
  const int lane = threadIdx.x & 31, wave = threadIdx.x >> 5;
  const int hi8 = ((lane >> 4) & 1) * 8;
#pragma unroll
  for (int t = wave; t < (COUT / 16) * NT; t += NWAVES) {
    int m = t / NT, n = t % NT;
    int xb = n * 16 + (lane & 15);
    int co0 = m * 16 + hi8;
    v8f acc;
    if (SCMODE == 1) {
      h8 xv = *(const h8*)(X + (size_t)(xb + 1) * CSTR + co0);
#pragma unroll
      for (int j = 0; j < 8; ++j) acc[j] = b2[co0 + j] + (float)xv[j];
    } else {
#pragma unroll
      for (int j = 0; j < 8; ++j) acc[j] = b2[co0 + j] + bs[co0 + j];
    }
    accum_gemm<KD32H>(acc, W2, H, xb + 1, m, lane);
    if (SCMODE == 2) accum_gemm<KD32X>(acc, Ws, X, xb + 1, m, lane);
    store_tile(D, acc, co0, xb);
  }
}

// Fused 6-ResBlock CNN stack over one spatial tile; whole stack stays in LDS.
// out[b][x] = stack(e)[x] for x in [0, NP)
__global__ __launch_bounds__(256) void roe_stack_kernel(const float* __restrict__ z,
                                                        StackArgs P,
                                                        float* __restrict__ out) {
  __shared__ _Float16 sA[NCOL * CSTR];
  __shared__ _Float16 sB[NCOL * CSTR];
  __shared__ _Float16 sC[NCOL * CSTR];
  const int tile = blockIdx.x, b = blockIdx.y;

  for (int i = threadIdx.x; i < NCOL * CSTR; i += 256) {
    sA[i] = (_Float16)0.f; sB[i] = (_Float16)0.f; sC[i] = (_Float16)0.f;
  }
  __syncthreads();

  // stage inputs into cols 1..128: ch0 = u[x-1], ch1 = u[x] (replicate clamp at global edges)
  const float* zb = z + (size_t)b * NZ;
  for (int p = threadIdx.x; p < WBUF; p += 256) {
    int ep = tile * TOUT - 5 + p;
    int x1 = clampi(ep, 0, NZ - 1);
    int x0 = clampi(ep - 1, 0, NZ - 1);
    sA[(size_t)(p + 1) * CSTR + 0] = (_Float16)zb[x0];
    sA[(size_t)(p + 1) * CSTR + 1] = (_Float16)zb[x1];
  }
  __syncthreads();

  // blk0: 2 -> 16, k3, conv shortcut
  conv3_pass<16, 32>(P.w1[0], P.b1[0], sA, sB);                       __syncthreads();
  resout_pass<16, 32, 2, 32>(P.w2[0], P.b2[0], sB, P.wsc[0], P.bsc[0], sA, sC); __syncthreads();
  // blk1: 16 -> 16, k3, identity
  conv3_pass<16, 32>(P.w1[1], P.b1[1], sC, sB);                       __syncthreads();
  resout_pass<16, 32, 1, 0>(P.w2[1], P.b2[1], sB, nullptr, nullptr, sC, sA);    __syncthreads();
  // blk2: 16 -> 32, k3, conv shortcut
  conv3_pass<32, 32>(P.w1[2], P.b1[2], sA, sB);                       __syncthreads();
  resout_pass<32, 32, 2, 32>(P.w2[2], P.b2[2], sB, P.wsc[1], P.bsc[1], sA, sC); __syncthreads();
  // blk3: 32 -> 64, k3, conv shortcut
  conv3_pass<64, 32>(P.w1[3], P.b1[3], sC, sB);                       __syncthreads();
  resout_pass<64, 64, 2, 32>(P.w2[3], P.b2[3], sB, P.wsc[2], P.bsc[2], sC, sA); __syncthreads();
  // blk4: 64 -> 64, k3, identity
  conv3_pass<64, 64>(P.w1[4], P.b1[4], sA, sB);                       __syncthreads();
  resout_pass<64, 64, 1, 0>(P.w2[4], P.b2[4], sB, nullptr, nullptr, sA, sC);    __syncthreads();
  // blk5: 64 -> 64, k1, identity
  conv1_pass<64, 64>(P.w1[5], P.b1[5], sC, sB);                       __syncthreads();
  resout_pass<64, 64, 1, 0>(P.w2[5], P.b2[5], sB, nullptr, nullptr, sC, sA);    __syncthreads();

  // final 1x1 conv 64 -> 1 (scalar VALU over contiguous channels)
  float fb0 = P.fb[0];
  for (int p = threadIdx.x; p < WBUF; p += 256) {
    if (p >= 5 && p < 5 + TOUT) {
      int ep = tile * TOUT + (p - 5);
      if (ep < NP) {
        float s = fb0;
        const _Float16* col = sA + (size_t)(p + 1) * CSTR;
#pragma unroll
        for (int ci = 0; ci < 64; ++ci) s += (float)P.fw[ci] * (float)col[ci];
        out[(size_t)b * NPAD + ep] = s;
      }
    }
  }
}

// Pointwise Roe combine + ghost-cell shuffle, fused:
// z_out[i] = z_new[src(i)], z_new[x] = u[x] + 0.5 * du[x]
__global__ __launch_bounds__(256) void roe_combine(const float* __restrict__ z,
                                                   const float* __restrict__ lam,
                                                   const float* __restrict__ L,
                                                   float* __restrict__ zo) {
  int i = blockIdx.x * blockDim.x + threadIdx.x;
  if (i >= NBATCH * NZ) return;
  int b = i / NZ, ix = i - b * NZ;
  int src = ix < 10 ? ix + 8192 : (ix < 8202 ? ix : ix - 8192);  // src in [10, 8201]
  const float* zb = z + (size_t)b * NZ;
  const float* lb = lam + (size_t)b * NPAD;
  const float* Lb = L + (size_t)b * NPAD;
  float um = zb[src], ul = zb[src - 1], ur = zb[src + 1];
  float laml = lb[src] * 0.1f, lamr = lb[src + 1] * 0.1f;
  float Ll = Lb[src], Lr = Lb[src + 1];
  float Rl = Ll / (Ll * Ll);     // inv(L^T L) @ L^T for 1x1
  float Rr = Lr / (Lr * Lr);
  float term_r = Rr * ((lamr - fabsf(lamr)) * (Lr * (ur - um)));
  float term_l = Rl * ((laml + fabsf(laml)) * (Ll * (um - ul)));
  float du = -(term_r + term_l) * 4096.0f;   // 1/(2*DX) = 4096
  zo[(size_t)b * NZ + ix] = um + 0.5f * du;  // h = Delta_t/steps = 0.5
}

// f32 -> f16 weight convert (1x1 convs): [Cout][Kd] -> [Cout][Kd32] zero-padded.
__global__ void roe_cvtw(const float* __restrict__ src, _Float16* __restrict__ dst,
                         int Kd, int Kd32, int n) {
  int i = blockIdx.x * blockDim.x + threadIdx.x;
  if (i >= n) return;
  int co = i / Kd32, k = i - co * Kd32;
  dst[i] = (k < Kd) ? (_Float16)src[co * Kd + k] : (_Float16)0.f;
}

// f32 -> f16 convert for k=3 convs: [Cout][Cin][3] -> [dx][Cout][Kd32(Cin)] zero-padded.
__global__ void roe_cvtw3(const float* __restrict__ src, _Float16* __restrict__ dst,
                          int Cin, int Cout, int Kd32, int n) {
  int i = blockIdx.x * blockDim.x + threadIdx.x;
  if (i >= n) return;
  int per = Cout * Kd32;
  int dx = i / per, r = i - dx * per;
  int co = r / Kd32, k = r - co * Kd32;
  dst[i] = (k < Cin) ? (_Float16)src[(co * Cin + k) * 3 + dx] : (_Float16)0.f;
}

// ---- host ----

// pytree leaf order (sorted dict keys) within one params stack:
// blk0{b1,b2,bs,w1,w2,ws} blk1{b1,b2,w1,w2} blk2{6} blk3{6} blk4{4} blk5{4} fb fw
static const int LEAF_SZ[32] = {
  16, 16, 16, 96,   256,  32,    // blk0
  16, 16, 768, 256,              // blk1
  32, 32, 32, 1536, 1024, 512,   // blk2
  64, 64, 64, 6144, 4096, 2048,  // blk3
  64, 64, 12288, 4096,           // blk4
  64, 64, 4096, 4096,            // blk5
  1, 64                          // fb, fw
};
static const int OFF_B1[6] = {0, 6, 10, 16, 22, 26};
static const int OFF_B2[6] = {1, 7, 11, 17, 23, 27};
static const int OFF_W1[6] = {3, 8, 13, 19, 24, 28};
static const int OFF_W2[6] = {4, 9, 14, 20, 25, 29};
static const int OFF_BS[3] = {2, 12, 18};
static const int OFF_WS[3] = {5, 15, 21};
static const int SC_BLK[3] = {0, 2, 3};

extern "C" void kernel_launch(void* const* d_in, const int* in_sizes, int n_in,
                              void* d_out, int out_size, void* d_ws, size_t ws_size,
                              hipStream_t stream) {
  (void)in_sizes; (void)out_size; (void)ws_size;
  const float* z0 = (const float*)d_in[0];

  const bool per_leaf = (n_in >= 66);
  auto leafPtr = [&](int stack, int li) -> const float* {
    if (per_leaf) return (const float*)d_in[1 + stack * 32 + li];
    const float* base = (const float*)d_in[1 + stack];
    long off = 0;
    for (int i = 0; i < li; ++i) off += LEAF_SZ[i];
    return base + off;
  };

  // workspace bump allocator
  char* wsb = (char*)d_ws;
  size_t pos = 0;
  auto alloc = [&](size_t bytes) -> void* {
    pos = (pos + 255) & ~(size_t)255;
    void* p = wsb + pos;
    pos += bytes;
    return p;
  };

  const int CI[6] = {2, 16, 16, 32, 64, 64};
  const int CO[6] = {16, 16, 32, 64, 64, 64};

  auto cvt1 = [&](const float* src, int Cout, int Kd) -> _Float16* {
    int Kd32 = (Kd + 31) & ~31;
    int n = Cout * Kd32;
    _Float16* dst = (_Float16*)alloc((size_t)n * sizeof(_Float16));
    roe_cvtw<<<(n + 255) / 256, 256, 0, stream>>>(src, dst, Kd, Kd32, n);
    return dst;
  };
  auto cvt3 = [&](const float* src, int Cout, int Cin) -> _Float16* {
    int Kd32 = (Cin + 31) & ~31;
    int n = 3 * Cout * Kd32;
    _Float16* dst = (_Float16*)alloc((size_t)n * sizeof(_Float16));
    roe_cvtw3<<<(n + 255) / 256, 256, 0, stream>>>(src, dst, Cin, Cout, Kd32, n);
    return dst;
  };

  StackArgs SP[2];
  for (int s = 0; s < 2; ++s) {
    StackArgs& A = SP[s];
    for (int blk = 0; blk < 6; ++blk) {
      if (blk < 5) A.w1[blk] = cvt3(leafPtr(s, OFF_W1[blk]), CO[blk], CI[blk]);
      else         A.w1[blk] = cvt1(leafPtr(s, OFF_W1[blk]), CO[blk], CI[blk]);
      A.w2[blk] = cvt1(leafPtr(s, OFF_W2[blk]), CO[blk], CO[blk]);
      A.b1[blk] = leafPtr(s, OFF_B1[blk]);
      A.b2[blk] = leafPtr(s, OFF_B2[blk]);
    }
    for (int k = 0; k < 3; ++k) {
      int blk = SC_BLK[k];
      A.wsc[k] = cvt1(leafPtr(s, OFF_WS[k]), CO[blk], CI[blk]);
      A.bsc[k] = leafPtr(s, OFF_BS[k]);
    }
    A.fw = cvt1(leafPtr(s, 31), 1, 64);
    A.fb = leafPtr(s, 30);
  }

  float* lam_arr = (float*)alloc((size_t)NBATCH * NPAD * sizeof(float));
  float* L_arr   = (float*)alloc((size_t)NBATCH * NPAD * sizeof(float));
  float* zmid    = (float*)alloc((size_t)NBATCH * NZ * sizeof(float));

  dim3 sgrid(NTILES, NBATCH);
  const int ncomb = NBATCH * NZ;
  for (int step = 0; step < 2; ++step) {   // steps = round(Delta_t/EPS) = 2
    const float* zin = (step == 0) ? z0 : zmid;
    float* zout = (step == 0) ? zmid : (float*)d_out;
    roe_stack_kernel<<<sgrid, 256, 0, stream>>>(zin, SP[0], lam_arr);
    roe_stack_kernel<<<sgrid, 256, 0, stream>>>(zin, SP[1], L_arr);
    roe_combine<<<(ncomb + 255) / 256, 256, 0, stream>>>(zin, lam_arr, L_arr, zout);
  }
}
